// GazeControlModel_63926293233728
// MI455X (gfx1250) — matrix operand order repaired and checked by reading, not compiled, not run on hardware
//
#include <hip/hip_runtime.h>
#include <hip/hip_bf16.h>
#include <math.h>

// ---------------------------------------------------------------------------
// GazeControlModel forward for MI455X (gfx1250, wave32, WMMA).
// f16 activations (NHWC, pow2-padded channels) + pre-packed f16 weight
// fragments -> v_wmma_f32_16x16x32_f16 with one 32B load per operand per
// K-tile. Irregular parts (gaussian scatter, pooling, heads) stay VALU.
// ---------------------------------------------------------------------------

typedef __attribute__((ext_vector_type(16))) _Float16 v16h;
typedef __attribute__((ext_vector_type(8)))  _Float16 v8h;
typedef __attribute__((ext_vector_type(8)))  float    v8f;

#define HM 32
#define WM 32
#define MCH 4
#define KNB 121
#define WVLD 496   // padded row stride for write_vals (484 -> 496, mult of 8)

__device__ __forceinline__ float act_relu(float x) { return x > 0.f ? x : 0.f; }
__device__ __forceinline__ float sigmoidf_(float x) { return 1.f / (1.f + __expf(-x)); }

// ---------------------------------------------------------------------------
// Weight repack into WMMA A-fragment order:
//   Wp[((mtile*kT + kt)*32 + lane)*16 + e]
// A-lane layout (16-bit A 16x32): row=l15, K = 16*h+e (contiguous)
// with h = lane>>4.
// mode 0: ksrc = ka                         (plain FC, Wsrc[OC][K])
// mode 1: conv OIHW, K-order (ky,kx,ic_pad): ic=ka%ICpad, w=ka/ICpad,
//         value = ic<ICsrc ? Wsrc[(oc*ICsrc+ic)*9+w] : 0
// mode 2: FC after NHWC conv flatten: c=ka%p0, s=ka/p0, ksrc=c*p1+s
// ---------------------------------------------------------------------------
__global__ void repack_w_kernel(const float* __restrict__ Wsrc, _Float16* __restrict__ Wp,
                                int OC, int K, int kT, int mode, int p0, int p1)
{
    int t = blockIdx.x * blockDim.x + threadIdx.x;
    int mtiles = (OC + 15) >> 4;
    int tot = mtiles * kT * 512;
    if (t >= tot) return;
    int e    = t & 15;
    int lane = (t >> 4) & 31;
    int kt   = (t >> 9) % kT;
    int mt   = (t >> 9) / kT;
    int h    = lane >> 4;
    int l15  = lane & 15;
    int arow = mt * 16 + l15;
    int ka   = kt * 32 + h * 16 + e;
    float v = 0.f;
    if (arow < OC && ka < K) {
        if (mode == 0) {
            v = Wsrc[(size_t)arow * K + ka];
        } else if (mode == 1) {
            int icp = p1;
            int w   = ka / icp;            // icp is pow2 at runtime; cheap enough here
            int ic  = ka - w * icp;
            if (ic < p0) v = Wsrc[((size_t)arow * p0 + ic) * 9 + w];
        } else {
            int c = ka % p0;
            int s = ka / p0;
            v = Wsrc[(size_t)arow * K + c * p1 + s];
        }
    }
    Wp[t] = (_Float16)v;
}

// ---------------------------------------------------------------------------
// FC GEMM: Out[row][feat] = act( X[row][0..K) . W[feat][0..K) + bias[feat] )
// WMMA-M = feature tile (pre-packed Wp), WMMA-N = batch-row tile.
// Requires: Nrows % 16 == 0, K % 32 == 0, ldX % 16 == 0, ldO % 8 == 0.
// ---------------------------------------------------------------------------
__global__ __launch_bounds__(128)
void gemm_wmma_f16(const _Float16* __restrict__ Wp, const float* __restrict__ bias,
                   const _Float16* __restrict__ X, int ldX,
                   _Float16* __restrict__ Out, int ldO,
                   int Nrows, int OC, int K, int act)
{
    const int lane = threadIdx.x & 31;
    const int wave = threadIdx.x >> 5;
    const int h    = lane >> 4;
    const int l15  = lane & 15;

    const int ntile = blockIdx.x * 4 + wave;
    const int mtile = blockIdx.y;
    if (ntile * 16 >= Nrows) return;

    const int kT   = K >> 5;
    const int ncol = ntile * 16 + l15;             // batch row (always valid)

    const _Float16* wp = Wp + ((size_t)mtile * kT * 32 + lane) * 16;
    const _Float16* xp = X + (size_t)ncol * ldX + h * 16;

    v8f acc = {};
    for (int kt = 0; kt < kT; ++kt) {
        v16h a = *(const v16h*)(wp + (size_t)kt * 512);
        v16h b = *(const v16h*)(xp + (size_t)kt * 32);
        acc = __builtin_amdgcn_wmma_f32_16x16x32_f16(
                  false, a, false, b, (short)0, acc, false, false);
    }

    const int m0 = mtile * 16 + h * 8;
    if (m0 + 8 <= OC) {
        v8h o;
#pragma unroll
        for (int r = 0; r < 8; ++r) {
            float v = acc[r] + bias[m0 + r];
            if (act == 1) v = act_relu(v);
            o[r] = (_Float16)v;
        }
        *(v8h*)(Out + (size_t)ncol * ldO + m0) = o;
    } else {
#pragma unroll
        for (int r = 0; r < 8; ++r) {
            if (m0 + r < OC) {
                float v = acc[r] + bias[m0 + r];
                if (act == 1) v = act_relu(v);
                Out[(size_t)ncol * ldO + m0 + r] = (_Float16)v;
            }
        }
    }
}

// ---------------------------------------------------------------------------
// Implicit-GEMM 3x3 conv on NHWC f16 with pow2 IC (channel-padded).
// K-order = (ky,kx,ic): each 16-half B fragment is a contiguous channel run
// at a single filter tap -> one 32B load + one uniform bounds test per ktile.
// Requires: Bn*OH*OW % 16 == 0, OH*OW / OW / IC powers of two, OC % 16 == 0.
// ---------------------------------------------------------------------------
__global__ __launch_bounds__(128)
void conv_wmma_f16(const _Float16* __restrict__ X, const _Float16* __restrict__ Wp,
                   const float* __restrict__ bias, _Float16* __restrict__ Out,
                   int Bn, int IH, int IW, int OC, int OH, int OW,
                   int stride, int pad, int act,
                   int icShift, int ohwShift, int owShift)
{
    const int lane = threadIdx.x & 31;
    const int wave = threadIdx.x >> 5;
    const int h    = lane >> 4;
    const int l15  = lane & 15;

    const int Ntot  = Bn << ohwShift;
    const int ntile = blockIdx.x * 4 + wave;
    const int mtile = blockIdx.y;
    if (ntile * 16 >= Ntot) return;

    const int IC = 1 << icShift;
    const int K  = IC * 9;
    const int kT = (K + 31) >> 5;

    const int ncol = ntile * 16 + l15;
    const int b  = ncol >> ohwShift;
    const int r_ = ncol & ((1 << ohwShift) - 1);
    const int oy = r_ >> owShift;
    const int ox = r_ & ((1 << owShift) - 1);
    const int iy0 = oy * stride - pad;
    const int ix0 = ox * stride - pad;

    const _Float16* wp = Wp + ((size_t)mtile * kT * 32 + lane) * 16;

    v8f acc = {};
    for (int kt = 0; kt < kT; ++kt) {
        v16h a = *(const v16h*)(wp + (size_t)kt * 512);
        const int kstart = kt * 32 + h * 16;
        v16h bv = {};
        if (kstart < K) {
            const int w   = kstart >> icShift;      // filter tap 0..8
            const int ic0 = kstart & (IC - 1);
            const int ky  = (w * 11) >> 5;          // w/3 for w in [0,8]
            const int kx  = w - ky * 3;
            const int iy  = iy0 + ky;
            const int ix  = ix0 + kx;
            if ((unsigned)iy < (unsigned)IH && (unsigned)ix < (unsigned)IW)
                bv = *(const v16h*)(X + (((size_t)(b * IH + iy) * IW + ix) << icShift) + ic0);
        }
        acc = __builtin_amdgcn_wmma_f32_16x16x32_f16(
                  false, a, false, bv, (short)0, acc, false, false);
    }

    const int m0 = mtile * 16 + h * 8;              // OC % 16 == 0 -> full store
    v8h o;
#pragma unroll
    for (int r = 0; r < 8; ++r) {
        float v = acc[r] + bias[m0 + r];
        if (act == 1) v = act_relu(v);
        o[r] = (_Float16)v;
    }
    *(v8h*)(Out + (size_t)ncol * OC + m0) = o;
}

// ---------------------------------------------------------------------------
// patch (B,3,16,16) f32 NCHW -> NHWC f16 padded to 16 channels.
// ---------------------------------------------------------------------------
__global__ void patch_cvt_kernel(const float* __restrict__ patch, _Float16* __restrict__ ph, int B)
{
    int t = blockIdx.x * blockDim.x + threadIdx.x;
    if (t >= B * 256 * 16) return;
    int ic  = t & 15;
    int pix = (t >> 4) & 255;
    int b   = t >> 12;
    float v = (ic < 3) ? patch[((size_t)b * 3 + ic) * 256 + pix] : 0.f;
    ph[t] = (_Float16)v;
}

// ---------------------------------------------------------------------------
// Positional encoding into featcat f16 columns 512..575.
// ---------------------------------------------------------------------------
__global__ void posenc_kernel(const float* __restrict__ gaze, _Float16* __restrict__ fc, int B)
{
    int b = blockIdx.x;
    int t = threadIdx.x;
    if (b >= B || t >= 64) return;
    int axis = t >> 5;
    int u    = t & 31;
    int j    = u >> 1;
    float d  = __expf(-(float)(2 * j) * logf(10000.f) / 32.f);
    float x  = gaze[b * 2 + axis] * d;
    float v  = (u & 1) ? __cosf(x) : __sinf(x);
    fc[(size_t)b * 576 + 512 + t] = (_Float16)v;
}

// ---------------------------------------------------------------------------
// Gaussian neighborhood geometry: idx[b][k], normalized weight nw[b][k].
// ---------------------------------------------------------------------------
__global__ __launch_bounds__(128)
void geometry_kernel(const float* __restrict__ gaze, int* __restrict__ idx,
                     float* __restrict__ nw, int B)
{
    __shared__ float sg[KNB];
    __shared__ float ssum;
    int b = blockIdx.x;
    int t = threadIdx.x;
    float gx = fminf(fmaxf(gaze[b * 2 + 0] * (WM - 1), 0.f), (float)(WM - 1));
    float gy = fminf(fmaxf(gaze[b * 2 + 1] * (HM - 1), 0.f), (float)(HM - 1));
    int x0 = (int)floorf(gx), y0 = (int)floorf(gy);
    int xs = 0, ys = 0;
    float g = 0.f;
    if (t < KNB) {
        int oxk = t % 11 - 5;          // meshgrid 'xy': ox fastest
        int oyk = t / 11 - 5;
        xs = min(max(x0 + oxk, 0), WM - 1);
        ys = min(max(y0 + oyk, 0), HM - 1);
        float dx = (float)xs - gx, dy = (float)ys - gy;
        const float sig = 11.f / 3.f;
        g = __expf(-(dx * dx + dy * dy) / (2.f * sig * sig));
        sg[t] = g;
    }
    __syncthreads();
    if (t == 0) {
        float s = 0.f;
        for (int i = 0; i < KNB; ++i) s += sg[i];
        ssum = fmaxf(s, 1e-8f);
    }
    __syncthreads();
    if (t < KNB) {
        nw[(size_t)b * KNB + t]  = g / ssum;
        idx[(size_t)b * KNB + t] = ys * WM + xs;
    }
}

// ---------------------------------------------------------------------------
// Per-cell strength gate MLP (8->64->1 sigmoid) + new_vals; emits wk, w*new.
// ---------------------------------------------------------------------------
__global__ __launch_bounds__(128)
void gate_kernel(const float* __restrict__ cell, const int* __restrict__ idx,
                 const float* __restrict__ nw, const _Float16* __restrict__ wvals,
                 const float* __restrict__ w1, const float* __restrict__ b1,
                 const float* __restrict__ w2, const float* __restrict__ b2,
                 float* __restrict__ wk, float* __restrict__ wnk, int B)
{
    int t = blockIdx.x * blockDim.x + threadIdx.x;
    if (t >= B * KNB) return;
    int b = t / KNB, k = t - b * KNB;
    int ii = idx[t];
    float gin[8];
#pragma unroll
    for (int m = 0; m < MCH; ++m) {
        gin[m]     = cell[((size_t)b * MCH + m) * (HM * WM) + ii];          // prev
        gin[4 + m] = (float)wvals[(size_t)b * WVLD + k * MCH + m];          // write val
    }
    float acc = b2[0];
    for (int j = 0; j < 64; ++j) {
        float hh = b1[j];
#pragma unroll
        for (int i = 0; i < 8; ++i) hh += w1[j * 8 + i] * gin[i];
        hh = act_relu(hh);
        acc += w2[j] * hh;
    }
    float w = nw[t] * sigmoidf_(acc);
    wk[t] = w;
#pragma unroll
    for (int m = 0; m < MCH; ++m) {
        float nv = (1.f - w) * gin[m] + w * gin[4 + m];
        wnk[(size_t)t * MCH + m] = w * nv;
    }
}

__global__ void fill_zero_kernel(float* p, size_t n)
{
    size_t i  = (size_t)blockIdx.x * blockDim.x + threadIdx.x;
    size_t st = (size_t)gridDim.x * blockDim.x;
    for (; i < n; i += st) p[i] = 0.f;
}

__global__ void scatter_kernel(const int* __restrict__ idx, const float* __restrict__ wk,
                               const float* __restrict__ wnk,
                               float* __restrict__ den, float* __restrict__ num, int B)
{
    int t = blockIdx.x * blockDim.x + threadIdx.x;
    if (t >= B * KNB) return;
    int b = t / KNB;
    int ii = idx[t];
    atomicAdd(&den[(size_t)b * (HM * WM) + ii], wk[t]);
#pragma unroll
    for (int m = 0; m < MCH; ++m)
        atomicAdd(&num[((size_t)b * (HM * WM) + ii) * MCH + m], wnk[(size_t)t * MCH + m]);
}

// Writes f32 NCHW `updated` (model output) + f16 NHWC 16-ch padded copy.
__global__ void update_kernel(const float* __restrict__ cell, const float* __restrict__ den,
                              const float* __restrict__ num, float* __restrict__ upd,
                              _Float16* __restrict__ updh, int B)
{
    int t = blockIdx.x * blockDim.x + threadIdx.x;
    if (t >= B * HM * WM) return;
    int b = t >> 10, hw = t & 1023;
    float keep = 1.f - fminf(den[t], 1.f);
    float vals[MCH];
#pragma unroll
    for (int m = 0; m < MCH; ++m) {
        float v = keep * cell[((size_t)b * MCH + m) * (HM * WM) + hw] + num[(size_t)t * MCH + m];
        upd[((size_t)b * MCH + m) * (HM * WM) + hw] = v;
        vals[m] = v;
    }
#pragma unroll
    for (int m = 0; m < 16; ++m)
        updh[(size_t)t * 16 + m] = (m < MCH) ? (_Float16)vals[m] : (_Float16)0.f;
}

// ---------------------------------------------------------------------------
// AdaptiveAvgPool2d(3), torch bin boundaries; input NHWC f16 (CB,32,32,64).
// Output pooled[b0+b][c*9+i*3+j] matching NCHW flatten of reference.
// ---------------------------------------------------------------------------
__global__ void apool_kernel(const _Float16* __restrict__ x, _Float16* __restrict__ pooled,
                             int CB, int b0)
{
    int t = blockIdx.x * blockDim.x + threadIdx.x;
    int total = CB * 64 * 9;
    if (t >= total) return;
    int b = t / (64 * 9);
    int r = t - b * (64 * 9);
    int c = r / 9;
    int ij = r - c * 9;
    int i = ij / 3, j = ij - i * 3;
    int si = (i * 32) / 3, ei = ((i + 1) * 32 + 2) / 3;
    int sj = (j * 32) / 3, ej = ((j + 1) * 32 + 2) / 3;
    float s = 0.f;
    for (int y = si; y < ei; ++y)
        for (int xx = sj; xx < ej; ++xx)
            s += (float)x[(((size_t)b * 32 + y) * 32 + xx) * 64 + c];
    pooled[(size_t)(b0 + b) * 576 + c * 9 + i * 3 + j] =
        (_Float16)(s / (float)((ei - si) * (ej - sj)));
}

// ---------------------------------------------------------------------------
// Agent heads from a (B,256) f16.
// ---------------------------------------------------------------------------
__global__ void heads_kernel(const _Float16* __restrict__ a,
                             const float* __restrict__ wgm, const float* __restrict__ bgm,
                             const float* __restrict__ wst, const float* __restrict__ bst,
                             const float* __restrict__ wv,  const float* __restrict__ bv,
                             float* __restrict__ out, int B,
                             int offStop, int offVal)
{
    int b = blockIdx.x * blockDim.x + threadIdx.x;
    if (b >= B) return;
    const _Float16* ar = a + (size_t)b * 256;
    float g0 = bgm[0], g1 = bgm[1], st = bst[0], vv = bv[0];
    for (int i = 0; i < 256; ++i) {
        float x = (float)ar[i];
        g0 += x * wgm[i];
        g1 += x * wgm[256 + i];
        st += x * wst[i];
        vv += x * wv[i];
    }
    out[b * 2 + 0]   = tanhf(g0);
    out[b * 2 + 1]   = tanhf(g1);
    out[offStop + b] = st;
    out[offVal + b]  = vv;
}

__global__ void std_kernel(const float* __restrict__ gls, float* __restrict__ out, int off)
{
    int t = threadIdx.x;
    if (t < 2) {
        float v = __expf(gls[t]);
        out[off + t] = fminf(fmaxf(v, 0.01f), 1.f);
    }
}

// ---------------------------------------------------------------------------
// Decoder final 1x1 conv (32->3) + sigmoid; input NHWC f16 (CB,32,32,32).
// ---------------------------------------------------------------------------
__global__ void dec1x1_kernel(const _Float16* __restrict__ d3, const float* __restrict__ w,
                              const float* __restrict__ bias, float* __restrict__ d4, int CB)
{
    int t = blockIdx.x * blockDim.x + threadIdx.x;
    if (t >= CB * HM * WM) return;
    int b = t >> 10, hw = t & 1023;
#pragma unroll
    for (int oc = 0; oc < 3; ++oc) {
        float acc = bias[oc];
        for (int ic = 0; ic < 32; ++ic)
            acc += (float)d3[((size_t)t) * 32 + ic] * w[oc * 32 + ic];
        d4[((size_t)b * 3 + oc) * (HM * WM) + hw] = sigmoidf_(acc);
    }
}

// ---------------------------------------------------------------------------
// Bilinear resize 32x32 -> 64x64 (half-pixel centers, clamp).
// ---------------------------------------------------------------------------
__global__ void resize_kernel(const float* __restrict__ d4, float* __restrict__ out,
                              int CB, int b0, int outOff)
{
    int t = blockIdx.x * blockDim.x + threadIdx.x;
    int total = CB * 3 * 64 * 64;
    if (t >= total) return;
    int b = t / (3 * 4096);
    int r = t - b * (3 * 4096);
    int c = r / 4096;
    int p = r - c * 4096;
    int oy = p >> 6, ox = p & 63;
    float fy = fminf(fmaxf((oy + 0.5f) * 0.5f - 0.5f, 0.f), 31.f);
    float fx = fminf(fmaxf((ox + 0.5f) * 0.5f - 0.5f, 0.f), 31.f);
    int y0 = (int)fy, x0 = (int)fx;
    int y1 = min(y0 + 1, 31), x1 = min(x0 + 1, 31);
    float wy = fy - (float)y0, wx = fx - (float)x0;
    const float* src = d4 + ((size_t)b * 3 + c) * (HM * WM);
    float v00 = src[y0 * 32 + x0], v01 = src[y0 * 32 + x1];
    float v10 = src[y1 * 32 + x0], v11 = src[y1 * 32 + x1];
    float v = v00 * (1 - wy) * (1 - wx) + v01 * (1 - wy) * wx
            + v10 * wy * (1 - wx)       + v11 * wy * wx;
    out[(size_t)outOff + (((size_t)(b0 + b) * 3 + c) * 64 + oy) * 64 + ox] = v;
}

// ---------------------------------------------------------------------------
// Host orchestration
// ---------------------------------------------------------------------------
static inline int ctz_(int v) { int s = 0; while (!(v & 1)) { v >>= 1; ++s; } return s; }

extern "C" void kernel_launch(void* const* d_in, const int* in_sizes, int n_in,
                              void* d_out, int out_size, void* d_ws, size_t ws_size,
                              hipStream_t stream)
{
    const float* patch = (const float*)d_in[0];
    const float* cell  = (const float*)d_in[1];
    const float* gaze  = (const float*)d_in[2];
    const float* const* P = (const float* const*)(d_in + 3);
    enum {
        e_w1 = 0, e_b1, e_w2, e_b2, e_w3, e_b3, e_w4, e_b4, e_fc_w, e_fc_b,
        ig_w1, ig_b1, ig_w2, ig_b2, wv_w, wv_b, s_w1, s_b1, s_w2, s_b2,
        og_cw1, og_cb1, og_cw2, og_cb2, og_fc_w, og_fc_b, f_w, f_b,
        a_sh_w, a_sh_b, a_gm_w, a_gm_b, a_gls, a_st_w, a_st_b, a_v_w, a_v_b,
        d_w1, d_b1, d_w2, d_b2, d_w3, d_b3, d_w4, d_b4
    };

    const int B = in_sizes[0] / (3 * 16 * 16);
    float* out = (float*)d_out;

    const int OFF_STD   = B * 2;
    const int OFF_STOP  = OFF_STD + 2;
    const int OFF_VAL   = OFF_STOP + B;
    const int OFF_RECON = OFF_VAL + B;
    const int OFF_UPD   = OFF_RECON + B * 3 * 64 * 64;
    float* upd = out + OFF_UPD;

    // ---- workspace carve-up (bytes, 256-aligned) ----
    char* wsB = (char*)d_ws;
    size_t off = 0;
    auto allocB = [&](size_t bytes) {
        bytes = (bytes + 255) & ~(size_t)255;
        char* p = wsB + off; off += bytes; return (void*)p;
    };
    auto allocH = [&](size_t n) { return (_Float16*)allocB(n * sizeof(_Float16)); };
    auto allocF = [&](size_t n) { return (float*)allocB(n * sizeof(float)); };

    const int CB = 64;   // batch chunk for the 32x32 maps
    _Float16* patchh  = allocH((size_t)B * 256 * 16);
    _Float16* x1h     = allocH((size_t)B * 64 * 32);
    _Float16* x2h     = allocH((size_t)B * 16 * 64);
    _Float16* x3h     = allocH((size_t)B * 4 * 128);
    _Float16* x4h     = allocH((size_t)B * 512);
    _Float16* featcat = allocH((size_t)B * 576);
    _Float16* g1      = allocH((size_t)B * 256);
    _Float16* g2      = allocH((size_t)B * 128);
    _Float16* wvals   = allocH((size_t)B * WVLD);
    _Float16* updh    = allocH((size_t)B * 1024 * 16);
    _Float16* pooled  = allocH((size_t)B * 576);
    _Float16* ogfc    = allocH((size_t)B * 576);
    _Float16* hbuf    = allocH((size_t)B * 512);
    _Float16* abuf    = allocH((size_t)B * 256);
    _Float16* chunkA  = allocH((size_t)CB * 1024 * 128);
    _Float16* chunkB  = allocH((size_t)CB * 1024 * 64);
    _Float16* chunkC  = allocH((size_t)CB * 1024 * 64);
    int*   idxbuf = (int*)allocB((size_t)B * KNB * 4);
    float* nwbuf  = allocF((size_t)B * KNB);
    float* wkbuf  = allocF((size_t)B * KNB);
    float* wnkbuf = allocF((size_t)B * KNB * MCH);
    float* den    = allocF((size_t)B * HM * WM);
    float* num    = allocF((size_t)B * HM * WM * MCH);   // contiguous after den
    float* d4buf  = allocF((size_t)CB * 3 * 1024);
    // packed weight fragments
    auto wpSize = [](int OC, int K) {
        return (size_t)((OC + 15) / 16) * ((K + 31) / 32) * 512;
    };
    _Float16* Wp_e1  = allocH(wpSize(32, 144));
    _Float16* Wp_e2  = allocH(wpSize(64, 288));
    _Float16* Wp_e3  = allocH(wpSize(128, 576));
    _Float16* Wp_e4  = allocH(wpSize(128, 1152));
    _Float16* Wp_efc = allocH(wpSize(512, 512));
    _Float16* Wp_ig1 = allocH(wpSize(256, 576));
    _Float16* Wp_ig2 = allocH(wpSize(128, 256));
    _Float16* Wp_wv  = allocH(wpSize(484, 128));
    _Float16* Wp_og1 = allocH(wpSize(128, 144));
    _Float16* Wp_og2 = allocH(wpSize(64, 1152));
    _Float16* Wp_ofc = allocH(wpSize(576, 576));
    _Float16* Wp_f   = allocH(wpSize(512, 576));
    _Float16* Wp_ash = allocH(wpSize(256, 512));
    _Float16* Wp_d1  = allocH(wpSize(64, 144));
    _Float16* Wp_d2  = allocH(wpSize(64, 576));
    _Float16* Wp_d3  = allocH(wpSize(32, 576));
    (void)ws_size; (void)n_in; (void)out_size;

    auto repack = [&](const float* Wsrc, _Float16* Wp, int OC, int K,
                      int mode, int p0, int p1) {
        int mt = (OC + 15) / 16, kT = (K + 31) / 32;
        int tot = mt * kT * 512;
        repack_w_kernel<<<(tot + 255) / 256, 256, 0, stream>>>(Wsrc, Wp, OC, K, kT,
                                                               mode, p0, p1);
    };
    auto launch_gemm = [&](const _Float16* Wp, const float* bias, const _Float16* X,
                           int ldX, _Float16* O, int ldO, int Nrows, int OC, int K,
                           int act) {
        int nt = Nrows / 16;
        dim3 grid((nt + 3) / 4, (OC + 15) / 16);
        gemm_wmma_f16<<<grid, 128, 0, stream>>>(Wp, bias, X, ldX, O, ldO,
                                                Nrows, OC, K, act);
    };
    auto launch_conv = [&](const _Float16* X, const _Float16* Wp, const float* bias,
                           _Float16* O, int Bn, int IC, int IH, int IW, int OC,
                           int s, int p, int act) {
        int OH = (IH + 2 * p - 3) / s + 1;
        int OW = (IW + 2 * p - 3) / s + 1;
        int Ntot = Bn * OH * OW;
        dim3 grid((Ntot / 16 + 3) / 4, OC / 16);
        conv_wmma_f16<<<grid, 128, 0, stream>>>(X, Wp, bias, O, Bn, IH, IW, OC, OH, OW,
                                                s, p, act, ctz_(IC), ctz_(OH * OW), ctz_(OW));
    };

    // ---- weight repack (cheap; deterministic per launch) ----
    repack(P[e_w1], Wp_e1, 32, 144, 1, 3, 16);
    repack(P[e_w2], Wp_e2, 64, 288, 1, 32, 32);
    repack(P[e_w3], Wp_e3, 128, 576, 1, 64, 64);
    repack(P[e_w4], Wp_e4, 128, 1152, 1, 128, 128);
    repack(P[e_fc_w], Wp_efc, 512, 512, 2, 128, 4);   // NHWC->NCHW flatten perm
    repack(P[ig_w1], Wp_ig1, 256, 576, 0, 0, 0);
    repack(P[ig_w2], Wp_ig2, 128, 256, 0, 0, 0);
    repack(P[wv_w],  Wp_wv, 484, 128, 0, 0, 0);
    repack(P[og_cw1], Wp_og1, 128, 144, 1, 4, 16);
    repack(P[og_cw2], Wp_og2, 64, 1152, 1, 128, 128);
    repack(P[og_fc_w], Wp_ofc, 576, 576, 0, 0, 0);
    repack(P[f_w], Wp_f, 512, 576, 0, 0, 0);
    repack(P[a_sh_w], Wp_ash, 256, 512, 0, 0, 0);
    repack(P[d_w1], Wp_d1, 64, 144, 1, 4, 16);
    repack(P[d_w2], Wp_d2, 64, 576, 1, 64, 64);
    repack(P[d_w3], Wp_d3, 32, 576, 1, 64, 64);

    // ---- encoder ----
    {
        int tot = B * 256 * 16;
        patch_cvt_kernel<<<(tot + 255) / 256, 256, 0, stream>>>(patch, patchh, B);
    }
    launch_conv(patchh, Wp_e1, P[e_b1], x1h, B, 16, 16, 16, 32, 2, 1, 1);
    launch_conv(x1h, Wp_e2, P[e_b2], x2h, B, 32, 8, 8, 64, 2, 1, 1);
    launch_conv(x2h, Wp_e3, P[e_b3], x3h, B, 64, 4, 4, 128, 2, 1, 1);
    launch_conv(x3h, Wp_e4, P[e_b4], x4h, B, 128, 2, 2, 128, 1, 1, 1);
    launch_gemm(Wp_efc, P[e_fc_b], x4h, 512, featcat, 576, B, 512, 512, 0);

    // ---- positional encoding + input-gate MLP + write values ----
    posenc_kernel<<<B, 64, 0, stream>>>(gaze, featcat, B);
    launch_gemm(Wp_ig1, P[ig_b1], featcat, 576, g1, 256, B, 256, 576, 1);
    launch_gemm(Wp_ig2, P[ig_b2], g1, 256, g2, 128, B, 128, 256, 1);
    launch_gemm(Wp_wv, P[wv_b], g2, 128, wvals, WVLD, B, 484, 128, 0);

    // ---- neighborhood geometry + gate + scatter + memory update ----
    geometry_kernel<<<B, 128, 0, stream>>>(gaze, idxbuf, nwbuf, B);
    {
        int tot = B * KNB;
        gate_kernel<<<(tot + 127) / 128, 128, 0, stream>>>(
            cell, idxbuf, nwbuf, wvals, P[s_w1], P[s_b1], P[s_w2], P[s_b2],
            wkbuf, wnkbuf, B);
        size_t zn = (size_t)B * HM * WM * (1 + MCH);
        fill_zero_kernel<<<1024, 256, 0, stream>>>(den, zn);
        scatter_kernel<<<(tot + 127) / 128, 128, 0, stream>>>(idxbuf, wkbuf, wnkbuf,
                                                              den, num, B);
        int ut = B * HM * WM;
        update_kernel<<<(ut + 255) / 256, 256, 0, stream>>>(cell, den, num, upd, updh, B);
    }

    // ---- output gate (chunked) ----
    for (int cb = 0; cb < B; cb += CB) {
        const _Float16* uin = updh + (size_t)cb * 1024 * 16;
        launch_conv(uin, Wp_og1, P[og_cb1], chunkA, CB, 16, 32, 32, 128, 1, 1, 1);
        launch_conv(chunkA, Wp_og2, P[og_cb2], chunkB, CB, 128, 32, 32, 64, 1, 1, 1);
        int pt = CB * 64 * 9;
        apool_kernel<<<(pt + 255) / 256, 256, 0, stream>>>(chunkB, pooled, CB, cb);
    }
    launch_gemm(Wp_ofc, P[og_fc_b], pooled, 576, ogfc, 576, B, 576, 576, 0);

    // ---- fuse + agent heads ----
    launch_gemm(Wp_f, P[f_b], ogfc, 576, hbuf, 512, B, 512, 576, 1);
    launch_gemm(Wp_ash, P[a_sh_b], hbuf, 512, abuf, 256, B, 256, 512, 1);
    heads_kernel<<<(B + 63) / 64, 64, 0, stream>>>(abuf, P[a_gm_w], P[a_gm_b],
                                                   P[a_st_w], P[a_st_b],
                                                   P[a_v_w], P[a_v_b],
                                                   out, B, OFF_STOP, OFF_VAL);
    std_kernel<<<1, 32, 0, stream>>>(P[a_gls], out, OFF_STD);

    // ---- decoder (chunked) ----
    for (int cb = 0; cb < B; cb += CB) {
        const _Float16* uin = updh + (size_t)cb * 1024 * 16;
        launch_conv(uin, Wp_d1, P[d_b1], chunkB, CB, 16, 32, 32, 64, 1, 1, 1);
        launch_conv(chunkB, Wp_d2, P[d_b2], chunkC, CB, 64, 32, 32, 64, 1, 1, 1);
        launch_conv(chunkC, Wp_d3, P[d_b3], chunkB, CB, 64, 32, 32, 32, 1, 1, 1);
        int t1 = CB * 1024;
        dec1x1_kernel<<<(t1 + 255) / 256, 256, 0, stream>>>(chunkB, P[d_w4], P[d_b4],
                                                            d4buf, CB);
        int t2 = CB * 3 * 4096;
        resize_kernel<<<(t2 + 255) / 256, 256, 0, stream>>>(d4buf, out, CB, cb, OFF_RECON);
    }
}
